// TransformerBlock_42786464203576
// MI455X (gfx1250) — matrix operand order repaired and checked by reading, not compiled, not run on hardware
//
#include <hip/hip_runtime.h>

// ---------------------------------------------------------------------------
// Transformer block (pre-norm, causal MHA + FFN) for gfx1250 (MI455X).
// All GEMMs + attention run on v_wmma_f32_16x16x32_f16 (f16 in, f32 acc).
// Weights pre-transposed to [N][K] f16 so every WMMA operand fragment is a
// contiguous 4-DWORD (b128-mergeable) load. V stored transposed per head for
// the P.V contraction. GEMM: 4x2 WMMA tiles per wave (64x32), 128x128/block.
// ---------------------------------------------------------------------------

typedef __attribute__((ext_vector_type(16))) _Float16 v16h;
typedef __attribute__((ext_vector_type(8)))  float    v8f;
typedef unsigned int u32;

#define D_MODEL   1024
#define NUM_HEADS 16
#define D_K       64
#define D_FF      4096
#define BB        4
#define SS        2048
#define MROWS     (BB * SS)          // 8192
#define LN_EPS    1e-5f

union frag16 { v16h h; u32 u[8]; };

static __device__ __forceinline__ v8f wmma16(const frag16& a, const frag16& b, v8f c) {
  // D = A(16x32 f16) * B(32x16 f16) + C(16x16 f32)
  return __builtin_amdgcn_wmma_f32_16x16x32_f16(
      /*neg_a=*/false, a.h, /*neg_b=*/false, b.h,
      /*c_mod=*/(short)0, c, /*reuse_a=*/false, /*reuse_b=*/false);
}

// A-fragment (16x32 f16): lane row = lane&15, K pairs across VGPRs, halves
// split K 0..7,16..23 / 8..15,24..31 -> element pairs contiguous at (k0 + akb).
static __device__ __forceinline__ void load_frag_rowK(const _Float16* rowp,
                                                      frag16& f) {
  const u32* p = (const u32*)rowp;
  #pragma unroll
  for (int v = 0; v < 8; ++v) f.u[v] = p[(v < 4) ? v : v + 4];
}

// B-fragment (32x16 f16): lane col = lane&15, same K order as A-fragment (halves
// 0..7/16..23 and 8..15/24..31) when stored K-contiguous per column (Wt / V^T / K).
static __device__ __forceinline__ void load_frag_colK(const _Float16* colp,
                                                      frag16& f) {
  const u32* p = (const u32*)colp;
  #pragma unroll
  for (int v = 0; v < 8; ++v) f.u[v] = p[(v < 4) ? v : v + 4];
}

// 16-lane (half-wave) reductions — wave32, xor masks < 16 stay in the half.
static __device__ __forceinline__ float rmax16(float x) {
  #pragma unroll
  for (int off = 1; off < 16; off <<= 1) x = fmaxf(x, __shfl_xor(x, off, 32));
  return x;
}
static __device__ __forceinline__ float rsum16(float x) {
  #pragma unroll
  for (int off = 1; off < 16; off <<= 1) x += __shfl_xor(x, off, 32);
  return x;
}

// ---------------------------------------------------------------------------
// fp32 [K][N] -> f16 [N][K] transpose (weights), 32x32 LDS tiles.
// ---------------------------------------------------------------------------
__global__ __launch_bounds__(256)
void cvt_transpose_f16(const float* __restrict__ in, _Float16* __restrict__ out,
                       int K, int N) {
  __shared__ float tile[32][33];
  const int kt = blockIdx.y * 32, nt = blockIdx.x * 32;
  #pragma unroll
  for (int r = threadIdx.y; r < 32; r += 8)
    tile[r][threadIdx.x] = in[(size_t)(kt + r) * N + nt + threadIdx.x];
  __syncthreads();
  #pragma unroll
  for (int r = threadIdx.y; r < 32; r += 8)
    out[(size_t)(nt + r) * K + kt + threadIdx.x] = (_Float16)tile[threadIdx.x][r];
}

// ---------------------------------------------------------------------------
// LayerNorm: fp32 in, f16 out. One 256-thread block per row (D_MODEL=1024).
// ---------------------------------------------------------------------------
__global__ __launch_bounds__(256)
void layernorm_f16(const float* __restrict__ x, const float* __restrict__ g,
                   const float* __restrict__ b, _Float16* __restrict__ out) {
  const int row = blockIdx.x;
  const int tid = threadIdx.x;
  const float* xr = x + (size_t)row * D_MODEL;
  float s = 0.f, s2 = 0.f;
  #pragma unroll
  for (int i = tid; i < D_MODEL; i += 256) {
    float v = xr[i];
    s += v; s2 += v * v;
  }
  __shared__ float sh[512];
  sh[tid] = s; sh[256 + tid] = s2;
  __syncthreads();
  for (int off = 128; off > 0; off >>= 1) {
    if (tid < off) { sh[tid] += sh[tid + off]; sh[256 + tid] += sh[256 + tid + off]; }
    __syncthreads();
  }
  const float mu   = sh[0]   * (1.f / D_MODEL);
  const float var  = sh[256] * (1.f / D_MODEL) - mu * mu;
  const float rstd = rsqrtf(var + LN_EPS);
  _Float16* orow = out + (size_t)row * D_MODEL;
  #pragma unroll
  for (int i = tid; i < D_MODEL; i += 256)
    orow[i] = (_Float16)((xr[i] - mu) * rstd * g[i] + b[i]);
}

// ---------------------------------------------------------------------------
// WMMA GEMM: C[M,N] = A[M,K](f16 row-major) * Wt[N,K](f16, pre-transposed) + bias
// Block = 256 threads = 8 waves, 2(M) x 4(N); wave tile 64x32 (4x2 WMMA).
// Block tile 128x128. EPI: 0 f16 | 1 relu f16 | 2 fp32 residual-add
//                        3 V^T per-head f16 | 4 f16 scaled by 1/sqrt(d_k)
// ---------------------------------------------------------------------------
template <int EPI>
__global__ __launch_bounds__(256)
void gemm_wmma(const _Float16* __restrict__ A, const _Float16* __restrict__ Wt,
               const float* __restrict__ bias, const float* __restrict__ res,
               float* __restrict__ outF, _Float16* __restrict__ outH,
               int N, int Kd) {
  const int lane = threadIdx.x & 31;
  const int w    = threadIdx.x >> 5;
  const int wm   = w >> 2;            // 0..1
  const int wn   = w & 3;             // 0..3
  const int m0   = blockIdx.y * 128 + wm * 64;
  const int n0   = blockIdx.x * 128 + wn * 32;
  const int nn   = lane & 15;
  const int hf   = lane >> 4;         // half-wave select
  const int akb  = hf ? 8 : 0;        // A-frag K base per half
  const int bkb  = hf ? 8 : 0;        // B-frag K base per half

  v8f acc[4][2];
  #pragma unroll
  for (int ti = 0; ti < 4; ++ti)
    #pragma unroll
    for (int tj = 0; tj < 2; ++tj)
      #pragma unroll
      for (int i = 0; i < 8; ++i) acc[ti][tj][i] = 0.f;

  const _Float16* ar[4];
  #pragma unroll
  for (int ti = 0; ti < 4; ++ti)
    ar[ti] = A + (size_t)(m0 + ti * 16 + nn) * Kd;         // A rows (lane&15)
  const _Float16* br0 = Wt + (size_t)(n0 + nn)      * Kd;  // B cols as Wt rows
  const _Float16* br1 = Wt + (size_t)(n0 + 16 + nn) * Kd;

  for (int k0 = 0; k0 < Kd; k0 += 32) {
    if (k0 + 32 < Kd) {                 // global_prefetch_b8 of next K tile
      #pragma unroll
      for (int ti = 0; ti < 4; ++ti) __builtin_prefetch(ar[ti] + k0 + 32, 0, 1);
      __builtin_prefetch(br0 + k0 + 32, 0, 1);
      __builtin_prefetch(br1 + k0 + 32, 0, 1);
    }
    frag16 a[4], b0, b1;
    #pragma unroll
    for (int ti = 0; ti < 4; ++ti) load_frag_rowK(ar[ti] + k0 + akb, a[ti]);
    load_frag_colK(br0 + k0 + bkb, b0);
    load_frag_colK(br1 + k0 + bkb, b1);
    #pragma unroll
    for (int ti = 0; ti < 4; ++ti) {
      acc[ti][0] = wmma16(a[ti], b0, acc[ti][0]);
      acc[ti][1] = wmma16(a[ti], b1, acc[ti][1]);
    }
  }

  #pragma unroll
  for (int ti = 0; ti < 4; ++ti)
    #pragma unroll
    for (int tj = 0; tj < 2; ++tj)
      #pragma unroll
      for (int v = 0; v < 8; ++v) {
        const int m = m0 + ti * 16 + v + hf * 8;
        const int n = n0 + tj * 16 + nn;
        const float cv = acc[ti][tj][v] + bias[n];
        if (EPI == 0) {
          outH[(size_t)m * N + n] = (_Float16)cv;
        } else if (EPI == 1) {
          outH[(size_t)m * N + n] = (_Float16)fmaxf(cv, 0.f);
        } else if (EPI == 2) {
          const size_t idx = (size_t)m * N + n;
          outF[idx] = res[idx] + cv;
        } else if (EPI == 3) {          // V^T per head: [b][h][dk][s]
          const int b  = m >> 11;       // m / SS
          const int sq = m & (SS - 1);
          const int h  = n >> 6;        // n / D_K
          const int dk = n & (D_K - 1);
          outH[(((size_t)b * NUM_HEADS + h) * D_K + dk) * SS + sq] = (_Float16)cv;
        } else {                        // Q scaled by 1/sqrt(d_k) = 0.125
          outH[(size_t)m * N + n] = (_Float16)(cv * 0.125f);
        }
      }
}

// ---------------------------------------------------------------------------
// Flash-style causal attention. One wave per (b, h, 16-row q tile).
// Q/K f16 [b*S + s][D_MODEL] (head cols at h*D_K, Q pre-scaled by 0.125).
// VT f16 [b][h][dk][s]. O f16 [b*S + s][D_MODEL].
// Key blocks of 32 -> 4 WMMAs scores + 4 WMMAs P.V (K=32).
// ---------------------------------------------------------------------------
__global__ __launch_bounds__(32)
void attn_flash(const _Float16* __restrict__ Q, const _Float16* __restrict__ K,
                const _Float16* __restrict__ VT, _Float16* __restrict__ O) {
  const int lane = threadIdx.x & 31;
  const int nn   = lane & 15;
  const int hf   = lane >> 4;
  const int q0   = blockIdx.x * 16;
  const int hd   = blockIdx.y;
  const int bt   = blockIdx.z;
  const size_t base   = (size_t)bt * SS * D_MODEL + (size_t)hd * D_K;
  const size_t vtbase = ((size_t)bt * NUM_HEADS + hd) * D_K * SS;

  const int akb = hf ? 8 : 0;
  const int bkb = hf ? 8 : 0;

  __shared__ __align__(16) _Float16 pt[16 * 32];

  // Q fragments: A(16x32) for dk 0..31 and 32..63 (already scaled by 0.125)
  frag16 aq0, aq1;
  {
    const _Float16* qrow = Q + base + (size_t)(q0 + nn) * D_MODEL;
    load_frag_rowK(qrow + akb, aq0);
    load_frag_rowK(qrow + 32 + akb, aq1);
  }

  float Mr[8], Lr[8];
  v8f acc[4];
  #pragma unroll
  for (int v = 0; v < 8; ++v) { Mr[v] = -1e30f; Lr[v] = 0.f; }
  #pragma unroll
  for (int c = 0; c < 4; ++c)
    #pragma unroll
    for (int v = 0; v < 8; ++v) acc[c][v] = 0.f;

  for (int j = 0; j < q0 + 16; j += 32) {     // causal: keys <= q0+15
    // ---- scores: s0 = Q.K^T for keys j..j+15, s1 for j+16..j+31
    v8f s0, s1;
    #pragma unroll
    for (int v = 0; v < 8; ++v) { s0[v] = 0.f; s1[v] = 0.f; }
    {
      frag16 b0, b1;
      const _Float16* krow = K + base + (size_t)(j + nn) * D_MODEL + bkb;
      load_frag_colK(krow, b0);
      load_frag_colK(krow + 32, b1);
      s0 = wmma16(aq0, b0, s0);
      s0 = wmma16(aq1, b1, s0);
    }
    {
      frag16 b0, b1;
      const _Float16* krow = K + base + (size_t)(j + 16 + nn) * D_MODEL + bkb;
      load_frag_colK(krow, b0);
      load_frag_colK(krow + 32, b1);
      s1 = wmma16(aq0, b0, s1);
      s1 = wmma16(aq1, b1, s1);
    }

    // ---- causal mask only on diagonal blocks (wave-uniform branch)
    if (j + 31 > q0) {
      #pragma unroll
      for (int v = 0; v < 8; ++v) {
        const int qg = q0 + v + hf * 8;       // C layout: M = v + 8*half
        if (j + nn > qg)      s0[v] = -1e30f;
        if (j + 16 + nn > qg) s1[v] = -1e30f;
      }
    }

    // ---- online softmax (f32)
    float p0a[8], p1a[8], fs[8];
    #pragma unroll
    for (int v = 0; v < 8; ++v) {
      const float mx = rmax16(fmaxf(s0[v], s1[v]));
      const float nm = fmaxf(Mr[v], mx);
      fs[v]  = __expf(Mr[v] - nm);
      Mr[v]  = nm;
      p0a[v] = __expf(s0[v] - nm);
      p1a[v] = __expf(s1[v] - nm);
      Lr[v]  = Lr[v] * fs[v] + rsum16(p0a[v] + p1a[v]);
    }
    #pragma unroll
    for (int c = 0; c < 4; ++c)
      #pragma unroll
      for (int v = 0; v < 8; ++v) acc[c][v] *= fs[v];

    // ---- P: C-layout -> LDS -> A-layout fragment (16x32 f16)
    __syncthreads();
    #pragma unroll
    for (int v = 0; v < 8; ++v) {
      const int m = v + hf * 8;
      pt[m * 32 + nn]      = (_Float16)p0a[v];
      pt[m * 32 + 16 + nn] = (_Float16)p1a[v];
    }
    __syncthreads();
    frag16 pa;
    load_frag_rowK(pt + nn * 32 + akb, pa);

    // ---- O += P(16x32) . V(32x64): V^T rows are key-contiguous
    #pragma unroll
    for (int c = 0; c < 4; ++c) {
      frag16 bv;
      load_frag_colK(VT + vtbase + (size_t)(16 * c + nn) * SS + j + bkb, bv);
      acc[c] = wmma16(pa, bv, acc[c]);
    }
  }

  // ---- normalize and store
  float inv[8];
  #pragma unroll
  for (int v = 0; v < 8; ++v) inv[v] = 1.f / Lr[v];
  #pragma unroll
  for (int c = 0; c < 4; ++c)
    #pragma unroll
    for (int v = 0; v < 8; ++v) {
      const size_t idx = base + (size_t)(q0 + v + hf * 8) * D_MODEL + 16 * c + nn;
      O[idx] = (_Float16)(acc[c][v] * inv[v]);
    }
}

// ---------------------------------------------------------------------------
// Host-side orchestration
// ---------------------------------------------------------------------------
extern "C" void kernel_launch(void* const* d_in, const int* in_sizes, int n_in,
                              void* d_out, int out_size, void* d_ws, size_t ws_size,
                              hipStream_t stream) {
  (void)in_sizes; (void)n_in; (void)out_size; (void)ws_size;

  const float* x    = (const float*)d_in[0];
  const float* Wq   = (const float*)d_in[1];
  const float* bq   = (const float*)d_in[2];
  const float* Wk   = (const float*)d_in[3];
  const float* bk   = (const float*)d_in[4];
  const float* Wv   = (const float*)d_in[5];
  const float* bv   = (const float*)d_in[6];
  const float* Wo   = (const float*)d_in[7];
  const float* bo   = (const float*)d_in[8];
  const float* W1   = (const float*)d_in[9];
  const float* b1   = (const float*)d_in[10];
  const float* W2   = (const float*)d_in[11];
  const float* b2   = (const float*)d_in[12];
  const float* ln1g = (const float*)d_in[13];
  const float* ln1b = (const float*)d_in[14];
  const float* ln2g = (const float*)d_in[15];
  const float* ln2b = (const float*)d_in[16];
  float* out = (float*)d_out;

  // workspace carve-up (256B aligned)
  char* ws = (char*)d_ws;
  size_t off = 0;
  auto alloc = [&](size_t bytes) -> void* {
    void* p = ws + off;
    off += (bytes + 255) & ~(size_t)255;
    return p;
  };
  const size_t MD = (size_t)MROWS * D_MODEL;
  const size_t MF = (size_t)MROWS * D_FF;
  _Float16* h1   = (_Float16*)alloc(MD * 2);
  _Float16* qf   = (_Float16*)alloc(MD * 2);
  _Float16* kf   = (_Float16*)alloc(MD * 2);
  _Float16* vt   = (_Float16*)alloc(MD * 2);     // V^T per head
  _Float16* ao   = (_Float16*)alloc(MD * 2);
  float*    x1   = (float*)   alloc(MD * 4);
  _Float16* h2   = (_Float16*)alloc(MD * 2);
  _Float16* ff1  = (_Float16*)alloc(MF * 2);
  _Float16* wq16 = (_Float16*)alloc((size_t)D_MODEL * D_MODEL * 2);  // [N][K]
  _Float16* wk16 = (_Float16*)alloc((size_t)D_MODEL * D_MODEL * 2);
  _Float16* wv16 = (_Float16*)alloc((size_t)D_MODEL * D_MODEL * 2);
  _Float16* wo16 = (_Float16*)alloc((size_t)D_MODEL * D_MODEL * 2);
  _Float16* w116 = (_Float16*)alloc((size_t)D_MODEL * D_FF * 2);     // [4096][1024]
  _Float16* w216 = (_Float16*)alloc((size_t)D_FF * D_MODEL * 2);     // [1024][4096]

  // weight convert + transpose: [K][N] fp32 -> [N][K] f16
  const dim3 tb(32, 8);
  cvt_transpose_f16<<<dim3(D_MODEL / 32, D_MODEL / 32), tb, 0, stream>>>(Wq, wq16, D_MODEL, D_MODEL);
  cvt_transpose_f16<<<dim3(D_MODEL / 32, D_MODEL / 32), tb, 0, stream>>>(Wk, wk16, D_MODEL, D_MODEL);
  cvt_transpose_f16<<<dim3(D_MODEL / 32, D_MODEL / 32), tb, 0, stream>>>(Wv, wv16, D_MODEL, D_MODEL);
  cvt_transpose_f16<<<dim3(D_MODEL / 32, D_MODEL / 32), tb, 0, stream>>>(Wo, wo16, D_MODEL, D_MODEL);
  cvt_transpose_f16<<<dim3(D_FF / 32,    D_MODEL / 32), tb, 0, stream>>>(W1, w116, D_MODEL, D_FF);
  cvt_transpose_f16<<<dim3(D_MODEL / 32, D_FF / 32),    tb, 0, stream>>>(W2, w216, D_FF, D_MODEL);

  // ln1(x) -> h1
  layernorm_f16<<<MROWS, 256, 0, stream>>>(x, ln1g, ln1b, h1);

  // Q,K,V projections (M x 1024 . 1024 x 1024); Q pre-scaled, V transposed
  const dim3 gqkv(D_MODEL / 128, MROWS / 128);
  gemm_wmma<4><<<gqkv, 256, 0, stream>>>(h1, wq16, bq, nullptr, nullptr, qf, D_MODEL, D_MODEL);
  gemm_wmma<0><<<gqkv, 256, 0, stream>>>(h1, wk16, bk, nullptr, nullptr, kf, D_MODEL, D_MODEL);
  gemm_wmma<3><<<gqkv, 256, 0, stream>>>(h1, wv16, bv, nullptr, nullptr, vt, D_MODEL, D_MODEL);

  // causal flash attention
  attn_flash<<<dim3(SS / 16, NUM_HEADS, BB), 32, 0, stream>>>(qf, kf, vt, ao);

  // out-proj + residual: x1 = x + ao.Wo + bo
  gemm_wmma<2><<<gqkv, 256, 0, stream>>>(ao, wo16, bo, x, x1, nullptr, D_MODEL, D_MODEL);

  // ln2(x1) -> h2
  layernorm_f16<<<MROWS, 256, 0, stream>>>(x1, ln2g, ln2b, h2);

  // FFN: ff1 = relu(h2.W1 + b1), out = x1 + ff1.W2 + b2
  gemm_wmma<1><<<dim3(D_FF / 128, MROWS / 128), 256, 0, stream>>>(
      h2, w116, b1, nullptr, nullptr, ff1, D_FF, D_MODEL);
  gemm_wmma<2><<<gqkv, 256, 0, stream>>>(ff1, w216, b2, x1, out, nullptr, D_MODEL, D_FF);
}
